// CifPredictorV2_30631706755487
// MI455X (gfx1250) — matrix operand order repaired and checked by reading, not compiled, not run on hardware
//
#include <hip/hip_runtime.h>
#include <hip/hip_bf16.h>
#include <math.h>

#define B_    16
#define T_    4096
#define D_    512
#define LOUT  512
#define K3    1536   // 3*D

typedef __attribute__((ext_vector_type(16))) __bf16 v16bf;
typedef __attribute__((ext_vector_type(8)))  __bf16 v8bf;
typedef __attribute__((ext_vector_type(8)))  float  v8f;
typedef __attribute__((ext_vector_type(4)))  unsigned int v4u;
typedef __attribute__((ext_vector_type(8)))  int v8i;
typedef __attribute__((ext_vector_type(4)))  int v4i;

// ---------------- ws layout (bytes) ----------------
#define WS_WTR   0u         // bf16[512*1536]  = 1572864 B
#define WS_ARAW  0x180000u  // f32 [B*T]       =  262144 B
#define WS_CUR   0x1C0000u  // f32 [B*T]       =  262144 B
#define WS_REM   0x200000u  // f32 [B*512]     =   32768 B
#define WS_FPOS  0x208000u  // i32 [B*512]     =   32768 B
#define WS_NF    0x210000u  // i32 [B]

// ---------------- kernel 0a: weight repack to bf16, B-fragment friendly ----
// Wtr[n][K] , K = k*512 + i  (so GEMM K index matches im2col of time halo)
__global__ __launch_bounds__(256)
void repack_w_kernel(const float* __restrict__ conv_w, __bf16* __restrict__ Wtr) {
  int idx = blockIdx.x * 256 + threadIdx.x;           // over 512*1536
  if (idx >= D_ * K3) return;
  int n  = idx / K3;
  int K  = idx - n * K3;
  int ko = K >> 9;            // 0..2
  int i  = K & 511;
  Wtr[idx] = (__bf16)conv_w[(size_t)n * K3 + i * 3 + ko];
}

// ---------------- kernel 0b: zero the pre-activation accumulator -----------
__global__ __launch_bounds__(256)
void zero_kernel(float* __restrict__ p, int n) {
  int idx = blockIdx.x * 256 + threadIdx.x;
  if (idx < n) p[idx] = 0.f;
}

// ---------- TDM descriptor helper: 2-D bf16 tile load into LDS -------------
// Loads tile [rows=tile1][cols=tile0] of a row-major bf16 tensor with row
// stride 'stride0' (elements), tile origin at 'gaddr', into LDS at 'lds_off'.
__device__ __forceinline__ void tdm_load_w_chunk(const __bf16* gaddr,
                                                 unsigned lds_off) {
  unsigned long long ga = (unsigned long long)(size_t)gaddr;
  v4u g0;
  g0[0] = 1u;                              // count=1, user descriptor
  g0[1] = lds_off;                         // lds_addr (bytes)
  g0[2] = (unsigned)(ga & 0xffffffffu);    // global_addr[31:0]
  g0[3] = (unsigned)((ga >> 32) & 0x1ffffffu) | (2u << 30); // addr[56:32], type=2

  const unsigned dim0 = K3, dim1 = D_;     // tensor extents (elements)
  const unsigned t0 = 32, t1 = 128;        // tile extents
  const unsigned long long s0 = K3;        // dim0 stride (elements)
  v8i g1;
  g1[0] = (int)(1u << 16);                 // wg_mask=0, data_size=1 (2B)
  g1[1] = (int)((dim0 & 0xffffu) << 16);   // atomic_addr=0 | tensor_dim0[15:0]
  g1[2] = (int)((dim0 >> 16) | ((dim1 & 0xffffu) << 16));
  g1[3] = (int)((dim1 >> 16) | (t0 << 16));            // tensor_dim1 hi | tile_dim0
  g1[4] = (int)(t1);                                   // tile_dim1 | tile_dim2=0
  g1[5] = (int)(s0 & 0xffffffffu);                     // stride0[31:0]
  g1[6] = (int)((s0 >> 32) & 0xffffu);                 // stride0 hi | stride1 lo=0
  g1[7] = 0;
  v4i g2 = {0, 0, 0, 0};
  v4i g3 = {0, 0, 0, 0};
  v8i g4 = {0, 0, 0, 0, 0, 0, 0, 0};
  __builtin_amdgcn_tensor_load_to_lds(g0, g1, g2, g3, g4, 0);
}

// ---------------- kernel 1: conv-as-GEMM with WMMA + fused out-proj --------
// grid: (T/64, B, 512/128), block 256 = 8 waves (4 M-waves x 2 N-waves)
__global__ __launch_bounds__(256)
void conv_wmma_kernel(const float* __restrict__ hidden,
                      const __bf16* __restrict__ Wtr,
                      const float* __restrict__ conv_b,
                      const float* __restrict__ out_w,
                      float* __restrict__ alphas_raw) {
  __shared__ __attribute__((aligned(32))) __bf16 ldsA[66 * 512]; // t halo rows
  __shared__ __attribute__((aligned(32))) __bf16 ldsW[2][128 * 32]; // dbl buf
  __shared__ float aTile[64];

  const int tid  = threadIdx.x;
  const int b    = blockIdx.y;
  const int t0   = blockIdx.x * 64;
  const int nBlk = blockIdx.z * 128;

  const int lane  = tid & 31;
  const int wave  = tid >> 5;
  const int waveM = wave & 3;   // 4 M-waves * 16 rows
  const int waveN = wave >> 2;  // 2 N-waves * 64 cols
  const int lh    = lane >> 4;  // 0/1 half-lane group
  const int ll    = lane & 15;

  // ---- kick off TDM load of weight chunk kc=0 (wave 0 only) ----
  if (wave == 0) {
    tdm_load_w_chunk(Wtr + (size_t)nBlk * K3 + 0,
                     (unsigned)(size_t)(__bf16*)&ldsW[0][0]);
  }

  // ---- stage A tile (rows t0-1 .. t0+64) into LDS as bf16 (overlaps TDM) --
  for (int e = tid; e < 66 * 128; e += 256) {
    int r  = e >> 7;
    int c4 = (e & 127) << 2;
    int t  = t0 - 1 + r;
    float4 v = make_float4(0.f, 0.f, 0.f, 0.f);
    if (t >= 0 && t < T_)
      v = *(const float4*)&hidden[((size_t)b * T_ + t) * D_ + c4];
    union { __bf16 h[4]; unsigned long long u; } pk;
    pk.h[0] = (__bf16)v.x; pk.h[1] = (__bf16)v.y;
    pk.h[2] = (__bf16)v.z; pk.h[3] = (__bf16)v.w;
    *(unsigned long long*)&ldsA[r * 512 + c4] = pk.u;   // one ds_store_b64
  }
  if (tid < 64) aTile[tid] = 0.f;
  if (wave == 0) __builtin_amdgcn_s_wait_tensorcnt(0);  // chunk 0 landed
  __syncthreads();

  v8f acc[4] = {};              // 16x64 per wave, four 16x16 subtiles
  const int mRow = waveM * 16 + ll;

  for (int kc = 0; kc < 48; ++kc) {
    const int k0 = kc * 32;
    const int ko = k0 >> 9;     // which tap (time halo offset)
    const int i0 = k0 & 511;
    const int buf = kc & 1;

    // prefetch next weight chunk into the other buffer (TDM, async)
    if (wave == 0 && kc + 1 < 48) {
      tdm_load_w_chunk(Wtr + (size_t)nBlk * K3 + (kc + 1) * 32,
                       (unsigned)(size_t)(__bf16*)&ldsW[buf ^ 1][0]);
    }

    // A fragment: lane<16 -> K [0..15]; lane>=16 -> K [16..31]
    const __bf16* ap = &ldsA[(mRow + ko) * 512 + i0 + lh * 16];
    v8bf a_lo = *(const v8bf*)(ap);
    v8bf a_hi = *(const v8bf*)(ap + 8);
    v16bf a;
#pragma unroll
    for (int e = 0; e < 8; ++e) { a[e] = a_lo[e]; a[8 + e] = a_hi[e]; }

    // B fragment from LDS: lane l -> col n = ll, K = k0 + lh*16 .. +15
    const __bf16* wb = &ldsW[buf][(waveN * 64 + ll) * 32 + lh * 16];
#pragma unroll
    for (int j = 0; j < 4; ++j) {
      v16bf bf = *(const v16bf*)(wb + j * 16 * 32);
      acc[j] = __builtin_amdgcn_wmma_f32_16x16x32_bf16(
          false, a, false, bf, (short)0, acc[j], false, false);
    }

    if (wave == 0 && kc + 1 < 48) __builtin_amdgcn_s_wait_tensorcnt(0);
    __syncthreads();            // next chunk visible; buf free for rewrite
  }

  // ---- fused epilogue: relu(acc + conv_b) dotted with out_w ----
  float pm[8];
#pragma unroll
  for (int r = 0; r < 8; ++r) pm[r] = 0.f;
#pragma unroll
  for (int j = 0; j < 4; ++j) {
    int n = nBlk + waveN * 64 + j * 16 + ll;
    float cb = conv_b[n];
    float ow = out_w[n];
#pragma unroll
    for (int r = 0; r < 8; ++r) {
      float y = acc[j][r] + cb;
      y = y > 0.f ? y : 0.f;
      pm[r] += y * ow;
    }
  }
  // D layout: VGPR r -> M = r + 8*lh, N = ll
#pragma unroll
  for (int r = 0; r < 8; ++r)
    atomicAdd(&aTile[waveM * 16 + r + 8 * lh], pm[r]);
  __syncthreads();
  if (tid < 64)
    atomicAdd(&alphas_raw[(size_t)b * T_ + t0 + tid], aTile[tid]);
}

// ---------------- kernel 2: sigmoid, token_num, scaled alphas --------------
__global__ __launch_bounds__(256)
void alpha_kernel(const float* __restrict__ araw, const float* __restrict__ mask,
                  const float* __restrict__ tlen, const float* __restrict__ out_b,
                  float* __restrict__ tok_out, float* __restrict__ alpha_out) {
  __shared__ float red[256];
  const int b = blockIdx.x, tid = threadIdx.x;
  const float ob = out_b[0];
  float s = 0.f;
  for (int t = tid; t < T_; t += 256) {
    float a  = araw[(size_t)b * T_ + t] + ob;
    float sg = 1.f / (1.f + __expf(-a));
    sg = fmaxf(sg, 0.f);                       // SMOOTH=1, NOISE=0
    sg *= mask[(size_t)b * T_ + t];
    s += sg;
  }
  red[tid] = s; __syncthreads();
  for (int off = 128; off > 0; off >>= 1) {
    if (tid < off) red[tid] += red[tid + off];
    __syncthreads();
  }
  const float tok = red[0];
  if (tid == 0) tok_out[b] = tok;
  const float scale = tlen[b] / tok;
  for (int t = tid; t < T_; t += 256) {
    float a  = araw[(size_t)b * T_ + t] + ob;
    float sg = 1.f / (1.f + __expf(-a));
    sg = fmaxf(sg, 0.f) * mask[(size_t)b * T_ + t];
    alpha_out[(size_t)b * T_ + t] = sg * scale;
  }
}

// ---------------- kernel 3: sequential fire scan (scalar, per batch) -------
__global__ void scan_kernel(const float* __restrict__ alphas,
                            float* __restrict__ peak_out, float* __restrict__ cur_ws,
                            float* __restrict__ rem_ws, int* __restrict__ fpos_ws,
                            int* __restrict__ nf_ws) {
  int b = threadIdx.x;
  if (b >= B_) return;
  float integrate = 0.f;
  int slot = 0;
  for (int t = 0; t < T_; ++t) {
    float alpha = alphas[(size_t)b * T_ + t];
    float dist  = 1.f - integrate;
    integrate  += alpha;
    float fires = integrate;
    peak_out[(size_t)b * T_ + t] = fires;
    bool fire = fires >= 1.0f;                 // THRESHOLD
    float cur = fire ? dist : alpha;
    cur_ws[(size_t)b * T_ + t] = cur;
    if (fire) {
      if (slot < LOUT) {
        fpos_ws[b * LOUT + slot] = t;
        rem_ws[b * LOUT + slot]  = alpha - cur;
      }
      ++slot;
      integrate = fires - 1.0f;
    }
  }
  nf_ws[b] = slot < LOUT ? slot : LOUT;
}

// ---------------- kernel 4: per-slot segment gather (no atomics) -----------
// grid (LOUT, B), block 256; each thread owns 2 of 512 dims
__global__ __launch_bounds__(256)
void gather_kernel(const float* __restrict__ hidden, const float* __restrict__ cur_ws,
                   const float* __restrict__ rem_ws, const int* __restrict__ fpos_ws,
                   const int* __restrict__ nf_ws, float* __restrict__ out) {
  const int s = blockIdx.x, b = blockIdx.y, tid = threadIdx.x;
  float a0 = 0.f, a1 = 0.f;
  if (s < nf_ws[b]) {
    int tEnd = fpos_ws[b * LOUT + s];
    int tStart = 0;
    if (s > 0) {
      int tp = fpos_ws[b * LOUT + s - 1];
      tStart = tp + 1;
      float r = rem_ws[b * LOUT + s - 1];     // carry-in from previous fire
      const float* h = &hidden[((size_t)b * T_ + tp) * D_];
      a0 = r * h[tid]; a1 = r * h[tid + 256];
    }
    for (int t = tStart; t <= tEnd; ++t) {
      float c = cur_ws[(size_t)b * T_ + t];
      const float* h = &hidden[((size_t)b * T_ + t) * D_];
      a0 += c * h[tid]; a1 += c * h[tid + 256];
    }
  }
  float* o = &out[((size_t)b * LOUT + s) * D_];
  o[tid] = a0; o[tid + 256] = a1;
}

// ---------------- launch ---------------------------------------------------
extern "C" void kernel_launch(void* const* d_in, const int* in_sizes, int n_in,
                              void* d_out, int out_size, void* d_ws, size_t ws_size,
                              hipStream_t stream) {
  const float* hidden = (const float*)d_in[0];
  const float* mask   = (const float*)d_in[1];
  const float* tlen   = (const float*)d_in[2];
  const float* conv_w = (const float*)d_in[3];
  const float* conv_b = (const float*)d_in[4];
  const float* out_w  = (const float*)d_in[5];
  const float* out_b  = (const float*)d_in[6];

  char* ws = (char*)d_ws;
  __bf16* Wtr    = (__bf16*)(ws + WS_WTR);
  float*  araw   = (float*)(ws + WS_ARAW);
  float*  cur_ws = (float*)(ws + WS_CUR);
  float*  rem_ws = (float*)(ws + WS_REM);
  int*    fpos   = (int*)  (ws + WS_FPOS);
  int*    nf     = (int*)  (ws + WS_NF);

  float* out      = (float*)d_out;
  float* acoustic = out;                       // 16*512*512
  float* tok_out  = out + (size_t)B_ * LOUT * D_;      // +16
  float* alphas   = tok_out + B_;                      // +B*T
  float* peak     = alphas + (size_t)B_ * T_;          // +B*T

  repack_w_kernel<<<(D_ * K3 + 255) / 256, 256, 0, stream>>>(conv_w, Wtr);
  zero_kernel<<<(B_ * T_ + 255) / 256, 256, 0, stream>>>(araw, B_ * T_);

  dim3 g1(T_ / 64, B_, D_ / 128);
  conv_wmma_kernel<<<g1, 256, 0, stream>>>(hidden, Wtr, conv_b, out_w, araw);

  alpha_kernel<<<B_, 256, 0, stream>>>(araw, mask, tlen, out_b, tok_out, alphas);

  scan_kernel<<<1, 32, 0, stream>>>(alphas, peak, cur_ws, rem_ws, fpos, nf);

  dim3 g4(LOUT, B_);
  gather_kernel<<<g4, 256, 0, stream>>>(hidden, cur_ws, rem_ws, fpos, nf, acoustic);
}
